// NAT_69956427317304
// MI455X (gfx1250) — matrix operand order repaired and checked by reading, not compiled, not run on hardware
//
#include <hip/hip_runtime.h>
#include <math.h>

// ---------------------------------------------------------------------------
// NAT forward for MI455X (gfx1250, wave32). Dense layers use
// v_wmma_f32_16x16x32_bf16 with 4x2 register blocking (64x32 output per wave,
// 8 WMMAs per 32-deep K step) and fused bias/residual/GELU/bf16 epilogue.
// Everything launched on `stream`; scratch carved out of d_ws; no hipMalloc.
// ---------------------------------------------------------------------------

typedef __attribute__((ext_vector_type(16))) __bf16 v16bf;
typedef __attribute__((ext_vector_type(8)))  __bf16 v8bf;
typedef __attribute__((ext_vector_type(8)))  float  v8f;

// ---------------- 3x3 conv, pad 1, NHWC activations, HWIO weights ----------
__global__ void k_conv3x3(const float* __restrict__ x, const float* __restrict__ w,
                          const float* __restrict__ bias, float* __restrict__ y,
                          int B, int Hin, int Win, int Cin,
                          int Hout, int Wout, int Cout, int stride)
{
  int idx = blockIdx.x * blockDim.x + threadIdx.x;
  int total = B * Hout * Wout * Cout;
  if (idx >= total) return;
  int co = idx % Cout;
  int wo = (idx / Cout) % Wout;
  int ho = (idx / (Cout * Wout)) % Hout;
  int b  = idx / (Cout * Wout * Hout);
  float acc = bias ? bias[co] : 0.f;
  for (int kh = 0; kh < 3; ++kh) {
    int hi = ho * stride + kh - 1;
    if (hi < 0 || hi >= Hin) continue;
    for (int kw = 0; kw < 3; ++kw) {
      int wi = wo * stride + kw - 1;
      if (wi < 0 || wi >= Win) continue;
      const float* xp = x + ((size_t)(b * Hin + hi) * Win + wi) * Cin;
      const float* wp = w + (size_t)(kh * 3 + kw) * Cin * Cout + co;
      for (int ci = 0; ci < Cin; ++ci)
        acc = fmaf(xp[ci], wp[(size_t)ci * Cout], acc);
    }
  }
  y[idx] = acc;
}

// ---------------- LayerNorm (one wave per row), optional f32 + bf16 outputs -
__global__ void k_layernorm(const float* __restrict__ x, const float* __restrict__ g,
                            const float* __restrict__ b, float* __restrict__ y,
                            __bf16* __restrict__ ybf, int N, int C)
{
  int row  = blockIdx.x * (blockDim.x >> 5) + (threadIdx.x >> 5);
  int lane = threadIdx.x & 31;
  if (row >= N) return;
  const float* xr = x + (size_t)row * C;
  float s = 0.f, s2 = 0.f;
  for (int c = lane; c < C; c += 32) { float v = xr[c]; s += v; s2 += v * v; }
  for (int o = 16; o > 0; o >>= 1) {
    s  += __shfl_xor(s,  o, 32);
    s2 += __shfl_xor(s2, o, 32);
  }
  float mu  = s / (float)C;
  float var = s2 / (float)C - mu * mu;
  float rs  = rsqrtf(var + 1e-5f);
  for (int c = lane; c < C; c += 32) {
    float v = (xr[c] - mu) * rs * g[c] + b[c];
    if (y)   y[(size_t)row * C + c]   = v;
    if (ybf) ybf[(size_t)row * C + c] = (__bf16)v;
  }
}

// ---------------- elementwise helpers --------------------------------------
__global__ void k_cvt_bf16(const float* __restrict__ x, __bf16* __restrict__ y, int n)
{
  int i = blockIdx.x * blockDim.x + threadIdx.x;
  if (i < n) y[i] = (__bf16)x[i];
}

// weight (K x N, row-major) -> bf16 transposed (N x K) for contiguous B loads
__global__ void k_wT_bf16(const float* __restrict__ w, __bf16* __restrict__ wt, int K, int N)
{
  int i = blockIdx.x * blockDim.x + threadIdx.x;
  if (i >= K * N) return;
  int k = i / N, n = i % N;
  wt[(size_t)n * K + k] = (__bf16)w[i];
}

__global__ void k_avgpool(const float* __restrict__ x, float* __restrict__ y,
                          int B, int HW, int C)
{
  int i = blockIdx.x * blockDim.x + threadIdx.x;
  if (i >= B * C) return;
  int c = i % C, b = i / C;
  float s = 0.f;
  for (int p = 0; p < HW; ++p) s += x[((size_t)b * HW + p) * C + c];
  y[i] = s / (float)HW;
}

// ---------------- NCHW -> NHWC transpose of the input image ----------------
__global__ void k_nchw2nhwc(const float* __restrict__ x, float* __restrict__ y,
                            int B, int C, int H, int W)
{
  int i = blockIdx.x * blockDim.x + threadIdx.x;
  int total = B * C * H * W;
  if (i >= total) return;
  int c = i % C;
  int w = (i / C) % W;
  int h = (i / (C * W)) % H;
  int b = i / (C * W * H);
  y[i] = x[((size_t)(b * C + c) * H + h) * W + w];
}

// ---------------- WMMA GEMM --------------------------------------------------
// C[N,M] = act( A[N,K](bf16) * Bt[M,K](bf16)^T + bias + res )
// One wave per 64x32 tile of C: 4 row-tiles x 2 col-tiles, 8 v8f accumulators,
// 8 WMMAs per K-step with A fragments reused x2 and B fragments reused x4.
// Fused epilogue: optional bias, optional residual add, optional exact GELU,
// optional f32 and/or bf16 stores. Bounds handled by clamped loads + guarded
// stores; no EXEC divergence before the WMMAs.
__global__ void k_gemm_bf16_wmma(const __bf16* __restrict__ A,
                                 const __bf16* __restrict__ Bt,
                                 const float* __restrict__ bias,
                                 const float* __restrict__ res,
                                 float* __restrict__ C,
                                 __bf16* __restrict__ Cbf,
                                 int gelu, int N, int K, int M)
{
  int lane = threadIdx.x & 31;
  int hh   = lane >> 4;       // half-wave selector
  int r    = lane & 15;
  int row0 = blockIdx.y * 64;
  int col0 = blockIdx.x * 32;

  const __bf16* ap[4];
#pragma unroll
  for (int t = 0; t < 4; ++t) {
    int ar = row0 + t * 16 + r;
    if (ar > N - 1) ar = N - 1;
    ap[t] = A + (size_t)ar * K + (hh << 3);     // A frag: K = h*8+e, +16
  }
  const __bf16* bp[2];
#pragma unroll
  for (int u = 0; u < 2; ++u) {
    int bc = col0 + u * 16 + r;
    if (bc > M - 1) bc = M - 1;
    bp[u] = Bt + (size_t)bc * K + (hh << 3);    // B frag: K = h*8+e, +16
  }

  v8f acc[4][2];
#pragma unroll
  for (int t = 0; t < 4; ++t)
#pragma unroll
    for (int u = 0; u < 2; ++u)
      acc[t][u] = (v8f){0.f, 0.f, 0.f, 0.f, 0.f, 0.f, 0.f, 0.f};

  for (int k0 = 0; k0 < K; k0 += 32) {
    v16bf av[4], bv[2];
#pragma unroll
    for (int t = 0; t < 4; ++t) {
      v8bf a0 = *(const v8bf*)(ap[t] + k0);
      v8bf a1 = *(const v8bf*)(ap[t] + k0 + 16);
#pragma unroll
      for (int e = 0; e < 8; ++e) { av[t][e] = a0[e]; av[t][e + 8] = a1[e]; }
    }
#pragma unroll
    for (int u = 0; u < 2; ++u) {
      v8bf b0 = *(const v8bf*)(bp[u] + k0);
      v8bf b1 = *(const v8bf*)(bp[u] + k0 + 16);
#pragma unroll
      for (int e = 0; e < 8; ++e) { bv[u][e] = b0[e]; bv[u][e + 8] = b1[e]; }
    }
#pragma unroll
    for (int t = 0; t < 4; ++t)
#pragma unroll
      for (int u = 0; u < 2; ++u)
        acc[t][u] = __builtin_amdgcn_wmma_f32_16x16x32_bf16(
            false, av[t], false, bv[u], (short)0, acc[t][u], false, false);
  }

  float bb[2];
#pragma unroll
  for (int u = 0; u < 2; ++u) {
    int bc = col0 + u * 16 + r;
    if (bc > M - 1) bc = M - 1;
    bb[u] = bias ? bias[bc] : 0.f;
  }
#pragma unroll
  for (int t = 0; t < 4; ++t) {
#pragma unroll
    for (int u = 0; u < 2; ++u) {
#pragma unroll
      for (int v = 0; v < 8; ++v) {
        int row = row0 + t * 16 + (hh << 3) + v;  // C layout: VGPR v -> M=h*8+v
        int col = col0 + u * 16 + r;              //           lane%16 -> N
        if (row < N && col < M) {
          size_t o = (size_t)row * M + col;
          float val = acc[t][u][v] + bb[u];
          if (res)  val += res[o];
          if (gelu) val = 0.5f * val * (1.f + erff(val * 0.70710678118654752f));
          if (C)    C[o] = val;
          if (Cbf)  Cbf[o] = (__bf16)val;
        }
      }
    }
  }
}

// ---------------- neighborhood attention (K=7, dh=32), thread per token*head
__global__ void k_nat_attn(const float* __restrict__ qkv, const float* __restrict__ rpb,
                           float* __restrict__ out, int B, int H, int W, int heads)
{
  const int DH = 32;
  int idx = blockIdx.x * blockDim.x + threadIdx.x;
  int total = B * H * W * heads;
  if (idx >= total) return;
  int h = idx % heads;
  int j = (idx / heads) % W;
  int i = (idx / (heads * W)) % H;
  int b = idx / (heads * W * H);
  int C = heads * DH;
  int si = i - 3; if (si < 0) si = 0; if (si > H - 7) si = H - 7;
  int sj = j - 3; if (sj < 0) sj = 0; if (sj > W - 7) sj = W - 7;
  const float scale = 0.17677669529663688f;  // 32^-0.5

  size_t base = ((size_t)(b * H + i) * W + j) * 3 * C + h * DH;
  float q[DH];
#pragma unroll
  for (int d = 0; d < DH; ++d) q[d] = qkv[base + d] * scale;

  float p[49];
  float mx = -3.4e38f;
  for (int a = 0; a < 7; ++a) {
    for (int c = 0; c < 7; ++c) {
      const float* kp = qkv + ((size_t)(b * H + (si + a)) * W + (sj + c)) * 3 * C
                            + C + h * DH;
      float s = 0.f;
#pragma unroll
      for (int d = 0; d < DH; ++d) s = fmaf(q[d], kp[d], s);
      int bi = a - (i - si) + 6;
      int bj = c - (j - sj) + 6;
      s += rpb[((size_t)h * 13 + bi) * 13 + bj];
      p[a * 7 + c] = s;
      mx = fmaxf(mx, s);
    }
  }
  float sum = 0.f;
#pragma unroll
  for (int a = 0; a < 49; ++a) { float e = __expf(p[a] - mx); p[a] = e; sum += e; }
  float inv = 1.f / sum;

  float o[DH];
#pragma unroll
  for (int d = 0; d < DH; ++d) o[d] = 0.f;
  for (int a = 0; a < 7; ++a) {
    for (int c = 0; c < 7; ++c) {
      const float* vp = qkv + ((size_t)(b * H + (si + a)) * W + (sj + c)) * 3 * C
                            + 2 * C + h * DH;
      float wgt = p[a * 7 + c] * inv;
#pragma unroll
      for (int d = 0; d < DH; ++d) o[d] = fmaf(wgt, vp[d], o[d]);
    }
  }
  float* op = out + ((size_t)(b * H + i) * W + j) * C + h * DH;
#pragma unroll
  for (int d = 0; d < DH; ++d) op[d] = o[d];
}

// ---------------------------------------------------------------------------
extern "C" void kernel_launch(void* const* d_in, const int* in_sizes, int n_in,
                              void* d_out, int out_size, void* d_ws, size_t ws_size,
                              hipStream_t stream)
{
  (void)in_sizes; (void)n_in; (void)out_size; (void)ws_size;
  const int B = 4;
  const int depths[4] = {2, 2, 4, 2};
  const int headsL[4] = {2, 4, 8, 16};

  // ---- input pointer walk (x first, then params flattened alphabetically) --
  int pi = 0;
  auto F = [&]() { return (const float*)d_in[pi++]; };

  const float* X      = F();       // (4,3,224,224) NCHW
  const float* head_b = F();
  const float* head_w = F();

  struct Blk { const float *fc1b,*fc1w,*fc2b,*fc2w,*n1b,*n1g,*n2b,*n2g,*pb,*pw,*qb,*qw,*rpb; };
  Blk blk[4][4];
  const float *dw[3], *dnb[3], *dng[3];
  for (int l = 0; l < 4; ++l) {
    for (int d = 0; d < depths[l]; ++d) {
      Blk& bp = blk[l][d];
      bp.fc1b = F(); bp.fc1w = F(); bp.fc2b = F(); bp.fc2w = F();
      bp.n1b  = F(); bp.n1g  = F(); bp.n2b  = F(); bp.n2g  = F();
      bp.pb   = F(); bp.pw   = F(); bp.qb   = F(); bp.qw   = F();
      bp.rpb  = F();
    }
    if (l < 3) { dw[l] = F(); dnb[l] = F(); dng[l] = F(); }
  }
  const float* fnb  = F(); const float* fng  = F();
  const float* tc1b = F(); const float* tc1w = F();
  const float* tc2b = F(); const float* tc2w = F();
  const float* tnb  = F(); const float* tng  = F();

  // ---- workspace bump allocator -------------------------------------------
  char* wsb = (char*)d_ws;
  size_t off = 0;
  auto alloc = [&](size_t bytes) -> void* {
    void* p = wsb + off;
    off += (bytes + 255) & ~(size_t)255;
    return p;
  };

  float*  bufA   = (float*)alloc((size_t)4 * 112 * 112 * 32 * 4);  // 6.4 MB
  float*  bufB   = (float*)alloc((size_t)4 * 56 * 56 * 64 * 4);    // 3.2 MB
  float*  nhwc   = (float*)alloc((size_t)4 * 224 * 224 * 3 * 4);   // 2.4 MB
  float*  qkvb   = (float*)alloc((size_t)4 * 56 * 56 * 192 * 4);   // 9.6 MB
  float*  attnb  = (float*)alloc((size_t)4 * 56 * 56 * 64 * 4);    // 3.2 MB
  __bf16* actbf  = (__bf16*)alloc((size_t)4 * 56 * 56 * 192 * 2);  // 4.8 MB
  __bf16* hidbf  = (__bf16*)alloc((size_t)4 * 56 * 56 * 192 * 2);  // 4.8 MB
  float*  pooled = (float*)alloc((size_t)4 * 512 * 4);
  __bf16* pooledbf = (__bf16*)alloc((size_t)4 * 512 * 2);

  // ---- stage all GEMM weights to bf16, transposed (N x K) -----------------
  auto stage = [&](const float* w, int K, int N) -> const __bf16* {
    __bf16* wt = (__bf16*)alloc((size_t)K * N * 2);
    int n = K * N;
    k_wT_bf16<<<dim3((n + 255) / 256), dim3(256), 0, stream>>>(w, wt, K, N);
    return wt;
  };
  const __bf16 *qkvw[4][4], *projw[4][4], *fc1wt[4][4], *fc2wt[4][4];
  for (int l = 0; l < 4; ++l) {
    int dim = 64 << l;
    for (int d = 0; d < depths[l]; ++d) {
      qkvw[l][d]  = stage(blk[l][d].qw,   dim,     3 * dim);
      projw[l][d] = stage(blk[l][d].pw,   dim,     dim);
      fc1wt[l][d] = stage(blk[l][d].fc1w, dim,     3 * dim);
      fc2wt[l][d] = stage(blk[l][d].fc2w, 3 * dim, dim);
    }
  }
  const __bf16* headw = stage(head_w, 512, 1000);

  auto ln = [&](const float* x, const float* g, const float* b,
                float* y, __bf16* ybf, int N, int C) {
    k_layernorm<<<dim3((N + 7) / 8), dim3(256), 0, stream>>>(x, g, b, y, ybf, N, C);
  };
  auto gemm = [&](const __bf16* A, const __bf16* Bt, const float* bias,
                  const float* res, float* Cp, __bf16* Cbf, int gelu,
                  int N, int K, int M) {
    k_gemm_bf16_wmma<<<dim3((M + 31) / 32, (N + 63) / 64), dim3(32), 0, stream>>>(
        A, Bt, bias, res, Cp, Cbf, gelu, N, K, M);
  };

  // ---- tokenizer ----------------------------------------------------------
  {
    int tt = 4 * 3 * 224 * 224;
    k_nchw2nhwc<<<dim3((tt + 255) / 256), dim3(256), 0, stream>>>(X, nhwc, 4, 3, 224, 224);
  }
  int t1 = 4 * 112 * 112 * 32;
  k_conv3x3<<<dim3((t1 + 255) / 256), dim3(256), 0, stream>>>(
      nhwc, tc1w, tc1b, bufA, B, 224, 224, 3, 112, 112, 32, 2);
  int t2 = 4 * 56 * 56 * 64;
  k_conv3x3<<<dim3((t2 + 255) / 256), dim3(256), 0, stream>>>(
      bufA, tc2w, tc2b, bufB, B, 112, 112, 32, 56, 56, 64, 2);
  ln(bufB, tng, tnb, bufA, nullptr, B * 56 * 56, 64);

  // ---- levels -------------------------------------------------------------
  float* cur = bufA;
  int Hl = 56, Wl = 56;
  for (int l = 0; l < 4; ++l) {
    int dim = 64 << l;
    int heads = headsL[l];
    int N = B * Hl * Wl;
    int nd = N * dim;
    for (int d = 0; d < depths[l]; ++d) {
      const Blk& bp = blk[l][d];
      // attention branch
      ln(cur, bp.n1g, bp.n1b, nullptr, actbf, N, dim);
      gemm(actbf, qkvw[l][d], bp.qb, nullptr, qkvb, nullptr, 0, N, dim, 3 * dim);
      int tot = N * heads;
      k_nat_attn<<<dim3((tot + 127) / 128), dim3(128), 0, stream>>>(
          qkvb, bp.rpb, attnb, B, Hl, Wl, heads);
      k_cvt_bf16<<<dim3((nd + 255) / 256), dim3(256), 0, stream>>>(attnb, actbf, nd);
      // proj with fused residual:  cur = cur + attn @ Wp + bp
      gemm(actbf, projw[l][d], bp.pb, cur, cur, nullptr, 0, N, dim, dim);
      // MLP branch
      ln(cur, bp.n2g, bp.n2b, nullptr, actbf, N, dim);
      // fc1 with fused exact GELU, bf16 output
      gemm(actbf, fc1wt[l][d], bp.fc1b, nullptr, nullptr, hidbf, 1, N, dim, 3 * dim);
      // fc2 with fused residual:  cur = cur + h @ W2 + b2
      gemm(hidbf, fc2wt[l][d], bp.fc2b, cur, cur, nullptr, 0, N, 3 * dim, dim);
    }
    if (l < 3) {
      int Ho = Hl / 2, Wo = Wl / 2, Co = dim * 2;
      int tot = B * Ho * Wo * Co;
      k_conv3x3<<<dim3((tot + 255) / 256), dim3(256), 0, stream>>>(
          cur, dw[l], nullptr, bufB, B, Hl, Wl, dim, Ho, Wo, Co, 2);
      ln(bufB, dng[l], dnb[l], cur, nullptr, B * Ho * Wo, Co);
      Hl = Ho; Wl = Wo;
    }
  }

  // ---- head ---------------------------------------------------------------
  int Nf = B * Hl * Wl;  // 4 * 49
  ln(cur, fng, fnb, bufB, nullptr, Nf, 512);
  k_avgpool<<<dim3((B * 512 + 255) / 256), dim3(256), 0, stream>>>(
      bufB, pooled, B, Hl * Wl, 512);
  k_cvt_bf16<<<dim3((B * 512 + 255) / 256), dim3(256), 0, stream>>>(
      pooled, pooledbf, B * 512);
  gemm(pooledbf, headw, head_b, nullptr, (float*)d_out, nullptr, 0, B, 512, 1000);
}
